// ParticleNetV2_19825569038775
// MI455X (gfx1250) — matrix operand order repaired and checked by reading, not compiled, not run on hardware
//
#include <hip/hip_runtime.h>
#include <hip/hip_bf16.h>

// ---------------------------------------------------------------------------
// ParticleNet-style GNN for MI455X (gfx1250), WMMA f16 path.
//
// v2: GEMM stages swizzled weights in LDS (shared by all 8 waves) and
// register-blocks 2 M-tiles per wave, so each B fragment (ds_load_b128 x2)
// feeds two v_wmma_f32_16x16x32_f16 ops. Global B traffic drops ~8x vs v1.
// ---------------------------------------------------------------------------

typedef _Float16 h16;
typedef __attribute__((ext_vector_type(16))) _Float16 v16h;
typedef __attribute__((ext_vector_type(8)))  float    v8f;

#define G_   1024
#define P_   64
#define K_   4
#define NF_  16
#define NGF_ 9
#define NH_  128
#define NC_  3
#define N_   (G_ * P_)          // 65536
#define E_   (N_ * K_)          // 262144
#define EPS_ 1e-5f

// ---------------------------------------------------------------------------
// Weight prep: fold BN into (scale, shift), convert W to f16 in the WMMA
// B-fragment swizzled layout:
//   for k-chunk c (32 rows), n-tile t (16 cols):
//     lane l = (n&15) + 16*(k&16 ? 1:0);  half i = k&15
//     Wswz[(((c*8 + t)*32 + l)*16) + i]
// Each lane then reads its 16 halves as two contiguous 16B loads.
// ---------------------------------------------------------------------------
__global__ void prep_layer_kernel(const float* __restrict__ W,   // Kd x 128 row-major
                                  const float* __restrict__ lb,  // linear bias [128]
                                  const float* __restrict__ bnw,
                                  const float* __restrict__ bnb,
                                  const float* __restrict__ rm,
                                  const float* __restrict__ rv,
                                  h16*  __restrict__ Wswz,
                                  float* __restrict__ epi,       // [bias|scale|shift] x128
                                  int Kd) {
  int tid = blockIdx.x * blockDim.x + threadIdx.x;
  int total = Kd * 128;
  if (tid < total) {
    int k = tid >> 7, n = tid & 127;
    int c = k >> 5, kin = k & 31;
    int t = n >> 4, nin = n & 15;
    int l = nin + ((kin & 16) ? 16 : 0);
    int i = kin & 15;
    Wswz[(((long)c * 8 + t) * 32 + l) * 16 + i] = (h16)W[tid];
  }
  if (tid < 128) {
    float s = bnw[tid] / sqrtf(rv[tid] + EPS_);
    epi[tid]       = lb[tid];
    epi[128 + tid] = s;
    epi[256 + tid] = bnb[tid] - rm[tid] * s;
  }
}

// ---------------------------------------------------------------------------
// WMMA MLP layer: out[m, 0..127] = epi( in[m,:] @ W )   (M multiple of 256)
// 256 threads = 8 waves. Weights staged in LDS; each wave owns 2 M-tiles
// (32 rows) x 8 N-tiles -> 16 WMMAs per 32-wide K-chunk.
// ---------------------------------------------------------------------------
__global__ __launch_bounds__(256)
void wmma_mlp_kernel(const h16* __restrict__ A, int Kd,
                     const h16* __restrict__ Wswz,
                     const float* __restrict__ epi,
                     h16* __restrict__ outH, float* __restrict__ outF,
                     int relu) {
  extern __shared__ h16 lw[];            // Kd*128 halves, swizzled layout

  // cooperative weight staging: Kd*256 bytes as 16B vectors
  {
    const uint4* src = (const uint4*)Wswz;
    uint4*       dst = (uint4*)lw;
    const int nvec = Kd * 16;            // (Kd*128*2)/16
    for (int i = threadIdx.x; i < nvec; i += 256) dst[i] = src[i];
  }
  __syncthreads();

  const int lane = threadIdx.x & 31;
  const int wave = threadIdx.x >> 5;
  const long mt0 = ((long)blockIdx.x * 8 + wave) * 2;  // first of 2 m-tiles
  const int  hi16 = lane >> 4;           // 0 | 1
  const int  lm   = lane & 15;

  v8f acc[2][8];
#pragma unroll
  for (int u = 0; u < 2; ++u)
#pragma unroll
    for (int t = 0; t < 8; ++t) acc[u][t] = (v8f){0.f,0.f,0.f,0.f,0.f,0.f,0.f,0.f};

  const int  KC  = Kd >> 5;
  const long rb0 = (mt0 * 16 + lm) * (long)Kd;
  const long rb1 = ((mt0 + 1) * 16 + lm) * (long)Kd;

  for (int c = 0; c < KC; ++c) {
    // A fragments (16x32 f16): lane = row (lane&15); k = c*32 + 16*hi16 + {0..15}
    union { v16h v; uint4 u[2]; } a0, a1;
    const h16* ap0 = A + rb0 + c * 32 + hi16 * 16;
    const h16* ap1 = A + rb1 + c * 32 + hi16 * 16;
    a0.u[0] = *(const uint4*)(ap0);
    a0.u[1] = *(const uint4*)(ap0 + 8);
    a1.u[0] = *(const uint4*)(ap1);
    a1.u[1] = *(const uint4*)(ap1 + 8);
#pragma unroll
    for (int t = 0; t < 8; ++t) {
      union { v16h v; uint4 u[2]; } bf;
      const h16* bp = lw + (((long)c * 8 + t) * 32 + lane) * 16;
      bf.u[0] = *(const uint4*)(bp);
      bf.u[1] = *(const uint4*)(bp + 8);
      acc[0][t] = __builtin_amdgcn_wmma_f32_16x16x32_f16(
          false, a0.v, false, bf.v, (short)0, acc[0][t], false, false);
      acc[1][t] = __builtin_amdgcn_wmma_f32_16x16x32_f16(
          false, a1.v, false, bf.v, (short)0, acc[1][t], false, false);
    }
  }

  const int rowOff = hi16 * 8;
#pragma unroll
  for (int t = 0; t < 8; ++t) {
    const int n = t * 16 + lm;
    const float bias  = epi[n];
    const float scale = epi[128 + n];
    const float shift = epi[256 + n];
#pragma unroll
    for (int u = 0; u < 2; ++u) {
#pragma unroll
      for (int r = 0; r < 8; ++r) {
        long m = (mt0 + u) * 16 + rowOff + r;
        float v = acc[u][t][r] + bias;
        if (relu) v = v > 0.f ? v : 0.f;
        v = v * scale + shift;
        if (outF) outF[m * 128 + n] = v;
        else      outH[m * 128 + n] = (h16)v;
      }
    }
  }
}

// ---------------------------------------------------------------------------
// GraphNorm: per group g (P=64 rows), per channel c:
//   out = (x - alpha*mean) / sqrt(var + eps) * w + b      (in-place safe)
// ---------------------------------------------------------------------------
__global__ void graphnorm_kernel(const float* __restrict__ in, float* __restrict__ out,
                                 const float* __restrict__ alpha,
                                 const float* __restrict__ w,
                                 const float* __restrict__ b, int C) {
  int g = blockIdx.x, c = threadIdx.x;
  if (c >= C) return;
  const float* base = in + (long)g * P_ * C + c;
  float s = 0.f;
  for (int p = 0; p < P_; ++p) s += base[(long)p * C];
  float am = alpha[c] * (s * (1.f / P_));
  float v = 0.f;
  for (int p = 0; p < P_; ++p) { float d = base[(long)p * C] - am; v += d * d; }
  float inv = (1.f / sqrtf(v * (1.f / P_) + EPS_)) * w[c];
  float bb = b[c];
  float* ob = out + (long)g * P_ * C + c;
  for (int p = 0; p < P_; ++p) ob[(long)p * C] = (base[(long)p * C] - am) * inv + bb;
}

// conv1 edge features: F[e] = [x_tgt (16) | x_src - x_tgt (16)] as f16
__global__ void build_edges1_kernel(const float* __restrict__ xg,
                                    const int* __restrict__ ei,
                                    h16* __restrict__ F) {
  long e = (long)blockIdx.x * 256 + threadIdx.x;
  if (e >= E_) return;
  int s = ei[e], t = ei[E_ + e];
  const float* xt = xg + (long)t * NF_;
  const float* xs = xg + (long)s * NF_;
  h16* f = F + e * 32;
#pragma unroll
  for (int c = 0; c < NF_; ++c) {
    float a = xt[c];
    f[c]        = (h16)a;
    f[NF_ + c]  = (h16)(xs[c] - a);
  }
}

// Per-group KNN (top-4 smallest squared distance, self excluded, ties -> lower idx)
__global__ __launch_bounds__(64)
void knn_kernel(const float* __restrict__ x, int* __restrict__ idx) {
  __shared__ float xs[P_ * NH_];
  __shared__ float sq[P_];
  int g = blockIdx.x, p = threadIdx.x;
  const float* row = x + ((long)g * P_ + p) * NH_;
  float s = 0.f;
  for (int c = 0; c < NH_; ++c) { float v = row[c]; xs[p * NH_ + c] = v; s += v * v; }
  sq[p] = s;
  __syncthreads();
  float bd[K_]; int bi[K_];
#pragma unroll
  for (int k = 0; k < K_; ++k) { bd[k] = 3.4e38f; bi[k] = 0; }
  const float* rp = xs + p * NH_;
  for (int q = 0; q < P_; ++q) {
    if (q == p) continue;
    const float* rq = xs + q * NH_;
    float dot = 0.f;
    for (int c = 0; c < NH_; ++c) dot += rp[c] * rq[c];
    float d = sq[p] + sq[q] - 2.f * dot;
    if (d < bd[K_ - 1]) {
      int k = K_ - 1;
      while (k > 0 && d < bd[k - 1]) { bd[k] = bd[k - 1]; bi[k] = bi[k - 1]; --k; }
      bd[k] = d; bi[k] = q;
    }
  }
  int* o = idx + ((long)g * P_ + p) * K_;
#pragma unroll
  for (int k = 0; k < K_; ++k) o[k] = g * P_ + bi[k];
}

// dyn conv edge features: F[e] = [xi (128) | xj - xi (128)] as f16
__global__ void build_edges2_kernel(const float* __restrict__ x,
                                    const int* __restrict__ idx,
                                    h16* __restrict__ F) {
  long tid = (long)blockIdx.x * 256 + threadIdx.x;   // E*32 threads, 4 cols each
  long e = tid >> 5;
  int  c0 = (int)(tid & 31) * 4;
  long i = e >> 2;
  int  j = idx[e];
  const float* xi = x + i * (long)NH_ + c0;
  const float* xj = x + (long)j * NH_ + c0;
  h16* f = F + e * 256;
#pragma unroll
  for (int c = 0; c < 4; ++c) {
    float a = xi[c];
    f[c0 + c]        = (h16)a;
    f[NH_ + c0 + c]  = (h16)(xj[c] - a);
  }
}

// conv1 combine: max over K msgs + inline 16->128 shortcut (lin + BN)
__global__ void combine1_kernel(const h16* __restrict__ msg, const float* __restrict__ xg,
                                const float* __restrict__ scw, const float* __restrict__ scb,
                                const float* __restrict__ bnw, const float* __restrict__ bnb,
                                const float* __restrict__ rm,  const float* __restrict__ rv,
                                float* __restrict__ out) {
  long tid = (long)blockIdx.x * 256 + threadIdx.x;   // N*128
  long i = tid >> 7; int n = tid & 127;
  const h16* mp = msg + i * K_ * NH_ + n;
  float m = -3.4e38f;
#pragma unroll
  for (int k = 0; k < K_; ++k) { float v = (float)mp[(long)k * NH_]; m = v > m ? v : m; }
  float acc = scb[n];
  const float* xr = xg + i * NF_;
#pragma unroll
  for (int k = 0; k < NF_; ++k) acc += xr[k] * scw[k * NH_ + n];
  float s = bnw[n] / sqrtf(rv[n] + EPS_);
  out[tid] = m + (acc - rm[n]) * s + bnb[n];
}

// dyn conv combine: max over K msgs + precomputed shortcut buffer
__global__ void combine2_kernel(const h16* __restrict__ msg, const float* __restrict__ sc,
                                float* __restrict__ out) {
  long tid = (long)blockIdx.x * 256 + threadIdx.x;   // N*128
  long i = tid >> 7; int n = tid & 127;
  const h16* mp = msg + i * K_ * NH_ + n;
  float m = -3.4e38f;
#pragma unroll
  for (int k = 0; k < K_; ++k) { float v = (float)mp[(long)k * NH_]; m = v > m ? v : m; }
  out[tid] = m + sc[tid];
}

__global__ void cvt_f16_kernel(const float* __restrict__ in, h16* __restrict__ out, long n) {
  long t = (long)blockIdx.x * 256 + threadIdx.x;
  if (t < n) out[t] = (h16)in[t];
}

// h[g][c] = mean over P of (c1+c2+c3)
__global__ void mean3_kernel(const float* __restrict__ a, const float* __restrict__ b,
                             const float* __restrict__ c, float* __restrict__ h) {
  int g = blockIdx.x, ch = threadIdx.x;   // 128 threads
  long base = (long)g * P_ * NH_ + ch;
  float s = 0.f;
  for (int p = 0; p < P_; ++p) { long o = base + (long)p * NH_; s += a[o] + b[o] + c[o]; }
  h[(long)g * NH_ + ch] = s * (1.f / P_);
}

// z = BN0(concat(h, graph_input)), 137 channels
__global__ void head_bn0_kernel(const float* __restrict__ h, const float* __restrict__ gi,
                                const float* __restrict__ b, const float* __restrict__ rm,
                                const float* __restrict__ rv, const float* __restrict__ w,
                                float* __restrict__ z) {
  int tid = blockIdx.x * blockDim.x + threadIdx.x;
  const int Cz = NH_ + NGF_;
  if (tid >= G_ * Cz) return;
  int g = tid / Cz, c = tid % Cz;
  float v = (c < NH_) ? h[(long)g * NH_ + c] : gi[(long)g * NGF_ + (c - NH_)];
  float s = w[c] / sqrtf(rv[c] + EPS_);
  z[tid] = (v - rm[c]) * s + b[c];
}

// Small dense head layer: out[g,n] = bn?( relu?( in[g,:]@W + b ) )
__global__ void head_lin_kernel(const float* __restrict__ in, int Kd,
                                const float* __restrict__ W, const float* __restrict__ b,
                                const float* __restrict__ bnw, const float* __restrict__ bnb,
                                const float* __restrict__ rm,  const float* __restrict__ rv,
                                int relu, float* __restrict__ out, int Nd) {
  int tid = blockIdx.x * blockDim.x + threadIdx.x;
  if (tid >= G_ * Nd) return;
  int g = tid / Nd, n = tid % Nd;
  float acc = b[n];
  const float* r = in + (long)g * Kd;
  for (int k = 0; k < Kd; ++k) acc += r[k] * W[k * Nd + n];
  if (relu) acc = acc > 0.f ? acc : 0.f;
  if (bnw) {
    float s = bnw[n] / sqrtf(rv[n] + EPS_);
    acc = (acc - rm[n]) * s + bnb[n];
  }
  out[tid] = acc;
}

// ---------------------------------------------------------------------------
// Host orchestration.
// Input order assumes jax pytree flattening (dict keys sorted alphabetically):
//  0:x 1:edge_index 2:graph_input
//  3-6:bn0{b,rm,rv,w} 7-10:bn1 11-14:bn2
//  conv1: 15-18 mlp.bn1{b,rm,rv,w} 19-22 mlp.bn2 23-26 mlp.bn3
//         27:l1.b 28:l1.w 29:l2.b 30:l2.w 31:l3.b 32:l3.w
//         33-36 sc_bn{b,rm,rv,w} 37:sc_lin.b 38:sc_lin.w
//  conv2: 39-62 (same layout, l1.w is 256x128)   conv3: 63-86
//  87:dense1.b 88:dense1.w 89:dense2.b 90:dense2.w
//  91-93:gn0{alpha,b,w} 94-96:gn1 97-99:gn2 100-102:gn3
//  103:output.b 104:output.w
// ---------------------------------------------------------------------------
extern "C" void kernel_launch(void* const* d_in, const int* in_sizes, int n_in,
                              void* d_out, int out_size, void* d_ws, size_t ws_size,
                              hipStream_t stream) {
  const float* x  = (const float*)d_in[0];
  const int*   ei = (const int*)d_in[1];
  const float* gi = (const float*)d_in[2];
  auto F32 = [&](int i) { return (const float*)d_in[i]; };

  // --- workspace carve-out ---
  char* ws = (char*)d_ws;
  size_t off = 0;
  auto carve = [&](size_t bytes) { size_t o = off; off = (off + bytes + 255) & ~(size_t)255; return ws + o; };
  float* XG   = (float*)carve((size_t)N_ * NF_ * 4);
  float* CN1  = (float*)carve((size_t)N_ * NH_ * 4);
  float* CN2  = (float*)carve((size_t)N_ * NH_ * 4);
  float* CN3  = (float*)carve((size_t)N_ * NH_ * 4);
  float* SC   = (float*)carve((size_t)N_ * NH_ * 4);
  h16*   CH   = (h16*)  carve((size_t)N_ * NH_ * 2);
  h16*   FBUF = (h16*)  carve((size_t)E_ * 256 * 2);   // edge features / GEMM pong
  h16*   MSGA = (h16*)  carve((size_t)E_ * NH_ * 2);   // GEMM ping
  int*   IDX  = (int*)  carve((size_t)E_ * 4);
  float* H    = (float*)carve((size_t)G_ * NH_ * 4);
  float* Z    = (float*)carve((size_t)G_ * (NH_ + NGF_) * 4);
  float* D1   = (float*)carve((size_t)G_ * NH_ * 4);
  float* D2   = (float*)carve((size_t)G_ * NH_ * 4);

  // --- WMMA layer table ---
  struct Layer { int wi, bi, bnbi, rmi, rvi, bnwi, Kd; };
  const Layer L[11] = {
    {28, 27, 15, 16, 17, 18,  32},  // 0 conv1.l1
    {30, 29, 19, 20, 21, 22, 128},  // 1 conv1.l2
    {32, 31, 23, 24, 25, 26, 128},  // 2 conv1.l3
    {52, 51, 39, 40, 41, 42, 256},  // 3 conv2.l1
    {54, 53, 43, 44, 45, 46, 128},  // 4 conv2.l2
    {56, 55, 47, 48, 49, 50, 128},  // 5 conv2.l3
    {62, 61, 57, 58, 59, 60, 128},  // 6 conv2.sc
    {76, 75, 63, 64, 65, 66, 256},  // 7 conv3.l1
    {78, 77, 67, 68, 69, 70, 128},  // 8 conv3.l2
    {80, 79, 71, 72, 73, 74, 128},  // 9 conv3.l3
    {86, 85, 81, 82, 83, 84, 128},  // 10 conv3.sc
  };
  h16*   WSWZ[11];
  float* EPI = (float*)carve(11 * 384 * 4);
  {
    size_t whalves = 0;
    for (int i = 0; i < 11; ++i) whalves += (size_t)L[i].Kd * 128;
    h16* wbase = (h16*)carve(whalves * 2);
    size_t o = 0;
    for (int i = 0; i < 11; ++i) { WSWZ[i] = wbase + o; o += (size_t)L[i].Kd * 128; }
  }
  for (int i = 0; i < 11; ++i) {
    int total = L[i].Kd * 128;
    prep_layer_kernel<<<(total + 255) / 256, 256, 0, stream>>>(
        F32(L[i].wi), F32(L[i].bi), F32(L[i].bnwi), F32(L[i].bnbi),
        F32(L[i].rmi), F32(L[i].rvi), WSWZ[i], EPI + (size_t)i * 384, L[i].Kd);
  }

  // GEMM launch: 256 rows per block, LDS = swizzled weights (Kd*256 bytes)
  auto gemm = [&](const h16* A, int layer, h16* oh, float* of, int relu, long M) {
    size_t shmem = (size_t)L[layer].Kd * 256;
    wmma_mlp_kernel<<<dim3((unsigned)(M / 256)), 256, shmem, stream>>>(
        A, L[layer].Kd, WSWZ[layer], EPI + (size_t)layer * 384, oh, of, relu);
  };

  const unsigned B_N128 = (unsigned)((long)N_ * NH_ / 256);   // 32768
  const unsigned B_E32  = (unsigned)((long)E_ * 32 / 256);    // 32768

  // ---- gn0 + conv1 (fixed edges) ----
  graphnorm_kernel<<<G_, NF_, 0, stream>>>(x, XG, F32(91), F32(93), F32(92), NF_);
  build_edges1_kernel<<<E_ / 256, 256, 0, stream>>>(XG, ei, FBUF);
  gemm(FBUF, 0, MSGA, nullptr, 1, E_);
  gemm(MSGA, 1, FBUF, nullptr, 1, E_);
  gemm(FBUF, 2, MSGA, nullptr, 1, E_);
  combine1_kernel<<<B_N128, 256, 0, stream>>>(MSGA, XG, F32(38), F32(37),
                                              F32(36), F32(33), F32(34), F32(35), CN1);
  graphnorm_kernel<<<G_, NH_, 0, stream>>>(CN1, CN1, F32(94), F32(96), F32(95), NH_);

  // ---- conv2 (dynamic KNN) ----
  cvt_f16_kernel<<<B_N128, 256, 0, stream>>>(CN1, CH, (long)N_ * NH_);
  knn_kernel<<<G_, P_, 0, stream>>>(CN1, IDX);
  build_edges2_kernel<<<B_E32, 256, 0, stream>>>(CN1, IDX, FBUF);
  gemm(FBUF, 3, MSGA, nullptr, 1, E_);
  gemm(MSGA, 4, FBUF, nullptr, 1, E_);
  gemm(FBUF, 5, MSGA, nullptr, 1, E_);
  gemm(CH,   6, nullptr, SC, 0, N_);
  combine2_kernel<<<B_N128, 256, 0, stream>>>(MSGA, SC, CN2);
  graphnorm_kernel<<<G_, NH_, 0, stream>>>(CN2, CN2, F32(97), F32(99), F32(98), NH_);

  // ---- conv3 ----
  cvt_f16_kernel<<<B_N128, 256, 0, stream>>>(CN2, CH, (long)N_ * NH_);
  knn_kernel<<<G_, P_, 0, stream>>>(CN2, IDX);
  build_edges2_kernel<<<B_E32, 256, 0, stream>>>(CN2, IDX, FBUF);
  gemm(FBUF, 7, MSGA, nullptr, 1, E_);
  gemm(MSGA, 8, FBUF, nullptr, 1, E_);
  gemm(FBUF, 9, MSGA, nullptr, 1, E_);
  gemm(CH,  10, nullptr, SC, 0, N_);
  combine2_kernel<<<B_N128, 256, 0, stream>>>(MSGA, SC, CN3);
  graphnorm_kernel<<<G_, NH_, 0, stream>>>(CN3, CN3, F32(100), F32(102), F32(101), NH_);

  // ---- head ----
  mean3_kernel<<<G_, NH_, 0, stream>>>(CN1, CN2, CN3, H);
  head_bn0_kernel<<<(G_ * (NH_ + NGF_) + 255) / 256, 256, 0, stream>>>(
      H, gi, F32(3), F32(4), F32(5), F32(6), Z);
  head_lin_kernel<<<(G_ * NH_ + 255) / 256, 256, 0, stream>>>(
      Z, NH_ + NGF_, F32(88), F32(87), F32(10), F32(7), F32(8), F32(9), 1, D1, NH_);
  head_lin_kernel<<<(G_ * NH_ + 255) / 256, 256, 0, stream>>>(
      D1, NH_, F32(90), F32(89), F32(14), F32(11), F32(12), F32(13), 1, D2, NH_);
  head_lin_kernel<<<(G_ * NC_ + 255) / 256, 256, 0, stream>>>(
      D2, NH_, F32(104), F32(103), nullptr, nullptr, nullptr, nullptr, 0,
      (float*)d_out, NC_);
}
